// PerformerHead_22187801051452
// MI455X (gfx1250) — matrix operand order repaired and checked by reading, not compiled, not run on hardware
//
#include <hip/hip_runtime.h>
#include <cstddef>
#include <cstdint>

// ---------------------------------------------------------------------------
// Performer forward (FAVOR+ causal linear attention), CDNA5 / gfx1250.
// GEMMs on v_wmma_f32_16x16x32_bf16; attention operands staged to LDS with
// async global->LDS copies (ASYNCcnt) when the toolchain exposes them.
// ---------------------------------------------------------------------------

typedef __bf16 bf16;
typedef __attribute__((ext_vector_type(16))) __bf16 v16bf;
typedef __attribute__((ext_vector_type(8)))  float  v8f;
typedef int vi4 __attribute__((vector_size(16)));   // b128 payload type

#define DEV __device__ __forceinline__
#define AS1 __attribute__((address_space(1)))
#define AS3 __attribute__((address_space(3)))

#if __has_builtin(__builtin_amdgcn_global_load_async_to_lds_b128)
#define HAS_ASYNC_LDS 1
#endif

constexpr int B_ = 16, N_ = 1024, DIM_ = 64, H_ = 4, DH_ = 64, L_ = 2, NC_ = 10;
constexpr int M_ = 266;          // FAVOR+ nb_features
constexpr int MP_ = 288;         // padded to multiple of 32 (WMMA K step)
constexpr int C_ = 128;          // causal chunk
constexpr int INNER_ = 256, NCH_ = 8;
constexpr int ROWS_ = B_ * N_;   // 16384
constexpr int BH_ = B_ * H_;     // 64
constexpr float DNORM_ = 0.35355339059327373f; // 64^{-1/4}
constexpr float RM_    = 0.06131393394849658f; // 266^{-1/2}

// ------------------------------ WMMA helpers -------------------------------

DEV v8f wmma_bf(v16bf a, v16bf b, v8f c) {
  return __builtin_amdgcn_wmma_f32_16x16x32_bf16(false, a, false, b, (short)0, c,
                                                 false, false);
}

// A fragment (16x32, element (r,k)) from row-major A, leading dim lda.
DEV v16bf load_a_rm(const bf16* A, int lda) {
  int l = threadIdx.x & 31;
  const bf16* p = A + (l & 15) * lda + ((l & 16) ? 8 : 0);
  v16bf a;
#pragma unroll
  for (int e = 0; e < 8; ++e) a[e] = p[e];
#pragma unroll
  for (int e = 8; e < 16; ++e) a[e] = p[e + 8];
  return a;
}

// A fragment where A(r,k) = T[k][r] (T row-major, leading dim ldt).
DEV v16bf load_at_rm(const bf16* T, int ldt) {
  int l = threadIdx.x & 31;
  const bf16* p = T + ((l & 16) ? 8 : 0) * ldt + (l & 15);
  v16bf a;
#pragma unroll
  for (int e = 0; e < 8; ++e) a[e] = p[e * ldt];
#pragma unroll
  for (int e = 8; e < 16; ++e) a[e] = p[(e + 8) * ldt];
  return a;
}

// B fragment (32x16, element (k,n)) from BT row-major [n][k] (contiguous in k).
DEV v16bf load_bt_rm(const bf16* BT, int ldbt) {
  int l = threadIdx.x & 31;
  const bf16* p = BT + (l & 15) * ldbt + ((l & 16) ? 8 : 0);
  v16bf b;
#pragma unroll
  for (int e = 0; e < 8; ++e) b[e] = p[e];
#pragma unroll
  for (int e = 8; e < 16; ++e) b[e] = p[e + 8];
  return b;
}

// B fragment from row-major B [k][n] (strided in k).
DEV v16bf load_b_rm(const bf16* Bm, int ldb) {
  int l = threadIdx.x & 31;
  const bf16* p = Bm + ((l & 16) ? 8 : 0) * ldb + (l & 15);
  v16bf b;
#pragma unroll
  for (int e = 0; e < 8; ++e) b[e] = p[e * ldb];
#pragma unroll
  for (int e = 8; e < 16; ++e) b[e] = p[(e + 8) * ldb];
  return b;
}

// ----------------------- async global -> LDS staging -----------------------

DEV void copy16_g2l(const bf16* g, bf16* l) {
#ifdef HAS_ASYNC_LDS
  __builtin_amdgcn_global_load_async_to_lds_b128((AS1 vi4*)g, (AS3 vi4*)l, 0, 0);
#else
  *(vi4*)(void*)l = *(const vi4*)(const void*)g;
#endif
}

DEV void wait_async() {
#ifdef HAS_ASYNC_LDS
#if __has_builtin(__builtin_amdgcn_s_wait_asynccnt)
  __builtin_amdgcn_s_wait_asynccnt(0);
#else
  asm volatile("s_wait_asynccnt 0x0" ::: "memory");
#endif
#endif
}

// ------------------------------ prep kernels -------------------------------

__global__ void k_embed(const int* __restrict__ x, const float* __restrict__ tok,
                        const float* __restrict__ pos, float* __restrict__ h) {
  int i = blockIdx.x * blockDim.x + threadIdx.x;
  if (i >= ROWS_ * DIM_) return;
  int d = i & 63;
  int n = (i >> 6) & (N_ - 1);
  int t = x[i >> 6];
  h[i] = tok[t * DIM_ + d] + pos[n * DIM_ + d];
}

__global__ void k_ln(const float* __restrict__ h, const float* __restrict__ g,
                     const float* __restrict__ b, bf16* __restrict__ y) {
  int r = blockIdx.x * blockDim.x + threadIdx.x;
  if (r >= ROWS_) return;
  const float* p = h + (size_t)r * DIM_;
  float m = 0.f;
  for (int d = 0; d < DIM_; ++d) m += p[d];
  m *= (1.f / DIM_);
  float v = 0.f;
  for (int d = 0; d < DIM_; ++d) { float t = p[d] - m; v += t * t; }
  v *= (1.f / DIM_);
  float inv = rsqrtf(v + 1e-5f);
  bf16* q = y + (size_t)r * DIM_;
  for (int d = 0; d < DIM_; ++d) q[d] = (bf16)((p[d] - m) * inv * g[d] + b[d]);
}

// W [K][Ncol] f32 -> WT [Ncol][K] bf16
__global__ void k_cvtT(const float* __restrict__ W, bf16* __restrict__ WT,
                       int K, int Ncol) {
  int i = blockIdx.x * blockDim.x + threadIdx.x;
  if (i >= K * Ncol) return;
  int k = i / Ncol, n = i % Ncol;
  WT[n * K + k] = (bf16)W[i];
}

// projS[m][d] = proj[m][d] * 64^{-1/4}, zero-padded m in [266,288)
__global__ void k_proj(const float* __restrict__ proj, bf16* __restrict__ projS) {
  int i = blockIdx.x * blockDim.x + threadIdx.x;
  if (i >= MP_ * DIM_) return;
  int m = i / DIM_, d = i % DIM_;
  projS[i] = (bf16)((m < M_) ? proj[m * DIM_ + d] * DNORM_ : 0.f);
}

// ------------------------------ GEMM kernel --------------------------------
// grid: (rows/16, NCOL/64), block 128 (4 waves); wave w -> 16x16 tile at col
// blockIdx.y*64 + w*16.  EPI: 0 = QKV bf16 scatter, 1 = GELU bf16, 2 = f32 +=
template <int EPI, int K>
__global__ __launch_bounds__(128)
void k_gemm(const bf16* __restrict__ A, const bf16* __restrict__ BT,
            const float* __restrict__ bias, bf16* __restrict__ outb,
            float* __restrict__ outf, int NCOL) {
  int rb = blockIdx.x * 16;
  int wave = threadIdx.x >> 5;
  int c0 = blockIdx.y * 64 + wave * 16;
  v8f acc = {};
#pragma unroll
  for (int kk = 0; kk < K; kk += 32)
    acc = wmma_bf(load_a_rm(A + (size_t)rb * K + kk, K),
                  load_bt_rm(BT + (size_t)c0 * K + kk, K), acc);
  int l = threadIdx.x & 31;
  int n = l & 15;
  int ro = (l & 16) ? 8 : 0;
#pragma unroll
  for (int v = 0; v < 8; ++v) {
    int row = rb + v + ro;
    int col = c0 + n;
    float val = acc[v] + bias[col];
    if (EPI == 0) {
      int bb = row >> 10, nn = row & (N_ - 1), hh = col >> 6, dh = col & 63;
      outb[(((size_t)(bb * H_ + hh) * N_) + nn) * DH_ + dh] = (bf16)val;
    } else if (EPI == 1) {
      float t = tanhf(0.7978845608028654f * (val + 0.044715f * val * val * val));
      outb[(size_t)row * NCOL + col] = (bf16)(0.5f * val * (1.f + t));
    } else {
      outf[(size_t)row * NCOL + col] += val;
    }
  }
}

// --------------------------- FAVOR+ feature maps ---------------------------

DEV void compute_dash(const bf16* qrow, const bf16* __restrict__ projS,
                      float dash[16][MP_]) {
  int wave = threadIdx.x >> 5;
  int l = threadIdx.x & 31;
  int n = l & 15;
  int ro = (l & 16) ? 8 : 0;
  v16bf a0 = load_a_rm(qrow, DIM_);
  v16bf a1 = load_a_rm(qrow + 32, DIM_);
  for (int t = wave; t < MP_ / 16; t += 4) {
    v8f acc = {};
    acc = wmma_bf(a0, load_bt_rm(projS + t * 16 * DIM_, DIM_), acc);
    acc = wmma_bf(a1, load_bt_rm(projS + t * 16 * DIM_ + 32, DIM_), acc);
#pragma unroll
    for (int v = 0; v < 8; ++v) dash[v + ro][t * 16 + n] = acc[v];
  }
}

__global__ __launch_bounds__(128)
void k_feat_q(const bf16* __restrict__ qb, const bf16* __restrict__ projS,
              bf16* __restrict__ qp) {
  __shared__ float dash[16][MP_];
  __shared__ float red[16];
  int rb = blockIdx.x * 16;
  compute_dash(qb + (size_t)rb * DIM_, projS, dash);
  __syncthreads();
  if (threadIdx.x < 16) {
    int r = threadIdx.x;
    const bf16* q = qb + (size_t)(rb + r) * DIM_;
    float ss = 0.f;
    for (int d = 0; d < DIM_; ++d) { float x = (float)q[d] * DNORM_; ss += x * x; }
    float mx = -1e30f;
    for (int m = 0; m < M_; ++m) mx = fmaxf(mx, dash[r][m]);
    red[r] = 0.5f * ss + mx;   // diag + per-row stab
  }
  __syncthreads();
  for (int i = threadIdx.x; i < 16 * MP_; i += 128) {
    int r = i / MP_, m = i % MP_;
    float val = (m < M_) ? (expf(dash[r][m] - red[r]) + 1e-4f) * RM_ : 0.f;
    qp[(size_t)(rb + r) * MP_ + m] = (bf16)val;
  }
}

__global__ __launch_bounds__(128)
void k_feat_kmax(const bf16* __restrict__ kb, const bf16* __restrict__ projS,
                 float* __restrict__ partial) {
  __shared__ float dash[16][MP_];
  __shared__ float smx[128];
  int rb = blockIdx.x * 16;
  compute_dash(kb + (size_t)rb * DIM_, projS, dash);
  __syncthreads();
  float mx = -1e30f;
  for (int i = threadIdx.x; i < 16 * MP_; i += 128) {
    int m = i % MP_;
    if (m < M_) mx = fmaxf(mx, dash[i / MP_][m]);
  }
  smx[threadIdx.x] = mx;
  __syncthreads();
  for (int s = 64; s; s >>= 1) {
    if (threadIdx.x < s) smx[threadIdx.x] = fmaxf(smx[threadIdx.x], smx[threadIdx.x + s]);
    __syncthreads();
  }
  if (threadIdx.x == 0) partial[blockIdx.x] = smx[0];
}

__global__ __launch_bounds__(256)
void k_rmax(const float* __restrict__ partial, int n, float* __restrict__ out) {
  __shared__ float smx[256];
  float mx = -1e30f;
  for (int i = threadIdx.x; i < n; i += 256) mx = fmaxf(mx, partial[i]);
  smx[threadIdx.x] = mx;
  __syncthreads();
  for (int s = 128; s; s >>= 1) {
    if (threadIdx.x < s) smx[threadIdx.x] = fmaxf(smx[threadIdx.x], smx[threadIdx.x + s]);
    __syncthreads();
  }
  if (threadIdx.x == 0) out[0] = smx[0];
}

__global__ __launch_bounds__(128)
void k_feat_k(const bf16* __restrict__ kb, const bf16* __restrict__ projS,
              const float* __restrict__ kmax, bf16* __restrict__ kp) {
  __shared__ float dash[16][MP_];
  __shared__ float red[16];
  int rb = blockIdx.x * 16;
  compute_dash(kb + (size_t)rb * DIM_, projS, dash);
  __syncthreads();
  if (threadIdx.x < 16) {
    int r = threadIdx.x;
    const bf16* q = kb + (size_t)(rb + r) * DIM_;
    float ss = 0.f;
    for (int d = 0; d < DIM_; ++d) { float x = (float)q[d] * DNORM_; ss += x * x; }
    red[r] = 0.5f * ss;
  }
  __syncthreads();
  float stab = kmax[0];
  for (int i = threadIdx.x; i < 16 * MP_; i += 128) {
    int r = i / MP_, m = i % MP_;
    float val = (m < M_) ? (expf(dash[r][m] - red[r] - stab) + 1e-4f) * RM_ : 0.f;
    kp[(size_t)(rb + r) * MP_ + m] = (bf16)val;
  }
}

// -------------------- per-chunk k^T v  (stored transposed) -----------------
// kvT[bh,ch][d][m] = sum_c v[c][d]*kp[c][m];  ksc[bh,ch][m] = sum_c kp[c][m]
__global__ __launch_bounds__(128)
void k_kv(const bf16* __restrict__ kp, const bf16* __restrict__ vb,
          bf16* __restrict__ kvT, float* __restrict__ ksc) {
  __shared__ __align__(16) bf16 Kls[C_][MP_];   // 72 KB
  __shared__ __align__(16) bf16 Vls[C_][DH_];   // 16 KB
  int bh = blockIdx.x >> 3, ch = blockIdx.x & 7;
  const bf16* kpc = kp + (size_t)(bh * N_ + ch * C_) * MP_;
  const bf16* vc  = vb + (size_t)(bh * N_ + ch * C_) * DH_;
  bf16* out = kvT + (size_t)blockIdx.x * DH_ * MP_;
  for (int i = threadIdx.x; i < (C_ * MP_) / 8; i += 128)
    copy16_g2l(kpc + (size_t)i * 8, &Kls[0][0] + (size_t)i * 8);
  for (int i = threadIdx.x; i < (C_ * DH_) / 8; i += 128)
    copy16_g2l(vc + (size_t)i * 8, &Vls[0][0] + (size_t)i * 8);
  wait_async();
  __syncthreads();
  int wave = threadIdx.x >> 5;
  int l = threadIdx.x & 31;
  int n = l & 15;
  int ro = (l & 16) ? 8 : 0;
  for (int t = wave; t < 4 * (MP_ / 16); t += 4) {
    int td = t / (MP_ / 16), tm = t % (MP_ / 16);
    v8f acc = {};
#pragma unroll
    for (int s = 0; s < C_ / 32; ++s)
      acc = wmma_bf(load_at_rm(&Vls[s * 32][td * 16], DH_),
                    load_b_rm(&Kls[s * 32][tm * 16], MP_), acc);
#pragma unroll
    for (int v = 0; v < 8; ++v)
      out[(size_t)(td * 16 + v + ro) * MP_ + tm * 16 + n] = (bf16)acc[v];
  }
  for (int m = threadIdx.x; m < MP_; m += 128) {
    float s = 0.f;
    for (int c = 0; c < C_; ++c) s += (float)Kls[c][m];
    ksc[(size_t)blockIdx.x * MP_ + m] = s;
  }
}

// Exclusive prefix over the 8 chunks (f32 accumulation).
__global__ void k_prefix_kv(const bf16* __restrict__ kvT, bf16* __restrict__ kvpT) {
  int i = blockIdx.x * blockDim.x + threadIdx.x;
  if (i >= BH_ * DH_ * MP_) return;
  int bh = i / (DH_ * MP_);
  int rem = i % (DH_ * MP_);
  size_t base = (size_t)bh * NCH_ * DH_ * MP_ + rem;
  float acc = 0.f;
  for (int c = 0; c < NCH_; ++c) {
    kvpT[base + (size_t)c * DH_ * MP_] = (bf16)acc;
    acc += (float)kvT[base + (size_t)c * DH_ * MP_];
  }
}

__global__ void k_prefix_ks(const float* __restrict__ ksc, float* __restrict__ ksp) {
  int i = blockIdx.x * blockDim.x + threadIdx.x;
  if (i >= BH_ * MP_) return;
  int bh = i / MP_, m = i % MP_;
  float acc = 0.f;
  for (int c = 0; c < NCH_; ++c) {
    ksp[(size_t)(bh * NCH_ + c) * MP_ + m] = acc;
    acc += ksc[(size_t)(bh * NCH_ + c) * MP_ + m];
  }
}

// ------------------- fused causal linear attention chunk -------------------
// out_i = (qp_i . kv_prefix + (masked S) v) / (qp_i . ks_prefix + rowsum(S))
// One 16-row tile per wave; q A-fragments cached in registers and reused for
// both the S GEMM and the num GEMM.  K/V staged in LDS via async copies.
__global__ __launch_bounds__(256)
void k_attn(const bf16* __restrict__ qp, const bf16* __restrict__ kp,
            const bf16* __restrict__ vb, const bf16* __restrict__ kvpT,
            const float* __restrict__ ksp, bf16* __restrict__ ob) {
  __shared__ __align__(16) bf16 Ks[C_][MP_];   // 72 KB
  __shared__ __align__(16) bf16 Vs[C_][DH_];   // 16 KB
  __shared__ __align__(16) bf16 S[C_][C_];     // 32 KB
  __shared__ float den[C_];
  int bh = blockIdx.x >> 3, ch = blockIdx.x & 7;
  int bb = bh >> 2, hh = bh & 3;
  const bf16* qc  = qp  + (size_t)(bh * N_ + ch * C_) * MP_;
  const bf16* kc  = kp  + (size_t)(bh * N_ + ch * C_) * MP_;
  const bf16* vc  = vb  + (size_t)(bh * N_ + ch * C_) * DH_;
  const bf16* kvp = kvpT + (size_t)(bh * NCH_ + ch) * DH_ * MP_;
  const float* ks = ksp + (size_t)(bh * NCH_ + ch) * MP_;
  int wave = threadIdx.x >> 5;
  int l = threadIdx.x & 31;
  int n = l & 15;
  int ro = (l & 16) ? 8 : 0;

  // async-stage K and V chunks into LDS
  for (int i = threadIdx.x; i < (C_ * MP_) / 8; i += 256)
    copy16_g2l(kc + (size_t)i * 8, &Ks[0][0] + (size_t)i * 8);
  for (int i = threadIdx.x; i < (C_ * DH_) / 8; i += 256)
    copy16_g2l(vc + (size_t)i * 8, &Vs[0][0] + (size_t)i * 8);

  // cache this wave's 16 q rows as 9 A-fragments (K = 288)
  int ti = wave;
  v16bf aq[MP_ / 32];
#pragma unroll
  for (int s = 0; s < MP_ / 32; ++s)
    aq[s] = load_a_rm(qc + (size_t)ti * 16 * MP_ + s * 32, MP_);

  wait_async();
  __syncthreads();

  // S = qp kp^T over feature dim, causal-masked, bf16 into LDS
  for (int tj = 0; tj < 8; ++tj) {
    v8f acc = {};
#pragma unroll
    for (int s = 0; s < MP_ / 32; ++s)
      acc = wmma_bf(aq[s], load_bt_rm(&Ks[tj * 16][s * 32], MP_), acc);
#pragma unroll
    for (int v = 0; v < 8; ++v) {
      int i = ti * 16 + v + ro, j = tj * 16 + n;
      S[i][j] = (bf16)((j <= i) ? acc[v] : 0.f);
    }
  }
  __syncthreads();

  // denominators
  if (threadIdx.x < C_) {
    int i = threadIdx.x;
    float d1 = 0.f;
    for (int m = 0; m < MP_; ++m) d1 += (float)qc[(size_t)i * MP_ + m] * ks[m];
    float d2 = 0.f;
    for (int j = 0; j <= i; ++j) d2 += (float)S[i][j];
    den[i] = d1 + d2;
  }
  __syncthreads();

  // num = qp * kv_prefix  +  S * v ; divide; scatter to [B,N,INNER]
  for (int td = 0; td < 4; ++td) {
    v8f acc = {};
#pragma unroll
    for (int s = 0; s < MP_ / 32; ++s)
      acc = wmma_bf(aq[s], load_bt_rm(kvp + (size_t)td * 16 * MP_ + s * 32, MP_), acc);
#pragma unroll
    for (int s = 0; s < C_ / 32; ++s)
      acc = wmma_bf(load_a_rm(&S[ti * 16][s * 32], C_),
                    load_b_rm(&Vs[s * 32][td * 16], DH_), acc);
#pragma unroll
    for (int v = 0; v < 8; ++v) {
      int i = ti * 16 + v + ro, d = td * 16 + n;
      int nn = ch * C_ + i;
      ob[((size_t)(bb * N_ + nn)) * INNER_ + hh * DH_ + d] = (bf16)(acc[v] / den[i]);
    }
  }
}

// ------------------------------ classifier ---------------------------------
__global__ __launch_bounds__(256)
void k_final(const float* __restrict__ h, const float* __restrict__ Wf,
             const float* __restrict__ bf_, float* __restrict__ out) {
  int bb = blockIdx.x / NC_, c = blockIdx.x % NC_;
  const float* hr = h + (size_t)bb * (N_ * DIM_);
  float s = 0.f;
  for (int i = threadIdx.x; i < N_ * DIM_; i += 256)
    s += hr[i] * Wf[(size_t)i * NC_ + c];
  __shared__ float red[256];
  red[threadIdx.x] = s;
  __syncthreads();
  for (int st = 128; st; st >>= 1) {
    if (threadIdx.x < st) red[threadIdx.x] += red[threadIdx.x + st];
    __syncthreads();
  }
  if (threadIdx.x == 0) out[bb * NC_ + c] = red[0] + bf_[c];
}

// ------------------------------ orchestration ------------------------------

extern "C" void kernel_launch(void* const* d_in, const int* in_sizes, int n_in,
                              void* d_out, int out_size, void* d_ws, size_t ws_size,
                              hipStream_t stream) {
  (void)in_sizes; (void)n_in; (void)out_size; (void)ws_size;
  const int*   x    = (const int*)  d_in[0];
  const float* tok  = (const float*)d_in[1];
  const float* pos  = (const float*)d_in[2];
  const float* proj = (const float*)d_in[3];
  const float* Wq   = (const float*)d_in[4];
  const float* bq   = (const float*)d_in[5];
  const float* Wk   = (const float*)d_in[6];
  const float* bk   = (const float*)d_in[7];
  const float* Wv   = (const float*)d_in[8];
  const float* bv   = (const float*)d_in[9];
  const float* Wo   = (const float*)d_in[10];
  const float* bo   = (const float*)d_in[11];
  const float* ln1g = (const float*)d_in[12];
  const float* ln1b = (const float*)d_in[13];
  const float* ln2g = (const float*)d_in[14];
  const float* ln2b = (const float*)d_in[15];
  const float* W1   = (const float*)d_in[16];
  const float* b1   = (const float*)d_in[17];
  const float* W2   = (const float*)d_in[18];
  const float* b2   = (const float*)d_in[19];
  const float* Wf   = (const float*)d_in[20];
  const float* bfv  = (const float*)d_in[21];

  constexpr size_t SZ_H    = (size_t)ROWS_ * DIM_ * 4;
  constexpr size_t SZ_Y    = (size_t)ROWS_ * DIM_ * 2;
  constexpr size_t SZ_QKV  = (size_t)BH_ * N_ * DH_ * 2;
  constexpr size_t SZ_FEAT = (size_t)BH_ * N_ * MP_ * 2;
  constexpr size_t SZ_KV   = (size_t)BH_ * NCH_ * DH_ * MP_ * 2;
  constexpr size_t SZ_KS   = (size_t)BH_ * NCH_ * MP_ * 4;
  constexpr size_t SZ_O    = (size_t)ROWS_ * INNER_ * 2;
  constexpr size_t WELEM   = (size_t)DIM_ * INNER_;       // 16384 per weight

  char* w = (char*)d_ws;
  float* hbuf   = (float*)w;            w += SZ_H;
  bf16*  ybuf   = (bf16*)w;             w += SZ_Y;
  bf16*  qbuf   = (bf16*)w;             w += SZ_QKV;
  bf16*  kbuf   = (bf16*)w;             w += SZ_QKV;
  bf16*  vbuf   = (bf16*)w;             w += SZ_QKV;
  bf16*  qpbuf  = (bf16*)w;             w += SZ_FEAT;
  bf16*  kpbuf  = (bf16*)w;             w += SZ_FEAT;
  bf16*  kvbuf  = (bf16*)w;             w += SZ_KV;
  bf16*  kvpbuf = (bf16*)w;             w += SZ_KV;
  float* kscbuf = (float*)w;            w += SZ_KS;
  float* kspbuf = (float*)w;            w += SZ_KS;
  bf16*  obuf   = (bf16*)w;             w += SZ_O;   // reused as GELU buffer
  bf16*  wts    = (bf16*)w;             w += (size_t)L_ * 6 * WELEM * 2;
  bf16*  projS  = (bf16*)w;             w += (size_t)MP_ * DIM_ * 2;
  float* partial= (float*)w;            w += 4096 * 4;
  float* kmax   = (float*)w;            w += 256;

  for (int l = 0; l < L_; ++l) {
    bf16* wl = wts + (size_t)l * 6 * WELEM;
    k_cvtT<<<64, 256, 0, stream>>>(Wq + (size_t)l * WELEM, wl + 0 * WELEM, 64, 256);
    k_cvtT<<<64, 256, 0, stream>>>(Wk + (size_t)l * WELEM, wl + 1 * WELEM, 64, 256);
    k_cvtT<<<64, 256, 0, stream>>>(Wv + (size_t)l * WELEM, wl + 2 * WELEM, 64, 256);
    k_cvtT<<<64, 256, 0, stream>>>(Wo + (size_t)l * WELEM, wl + 3 * WELEM, 256, 64);
    k_cvtT<<<64, 256, 0, stream>>>(W1 + (size_t)l * WELEM, wl + 4 * WELEM, 64, 256);
    k_cvtT<<<64, 256, 0, stream>>>(W2 + (size_t)l * WELEM, wl + 5 * WELEM, 256, 64);
  }
  k_proj<<<(MP_ * DIM_ + 255) / 256, 256, 0, stream>>>(proj, projS);
  k_embed<<<(ROWS_ * DIM_ + 255) / 256, 256, 0, stream>>>(x, tok, pos, hbuf);

  for (int l = 0; l < L_; ++l) {
    bf16* wqT = wts + (size_t)l * 6 * WELEM;
    bf16* wkT = wqT + WELEM;
    bf16* wvT = wqT + 2 * WELEM;
    bf16* woT = wqT + 3 * WELEM;
    bf16* w1T = wqT + 4 * WELEM;
    bf16* w2T = wqT + 5 * WELEM;

    k_ln<<<ROWS_ / 256, 256, 0, stream>>>(hbuf, ln1g + l * DIM_, ln1b + l * DIM_, ybuf);
    dim3 g4(ROWS_ / 16, 4), g1(ROWS_ / 16, 1);
    k_gemm<0, 64><<<g4, 128, 0, stream>>>(ybuf, wqT, bq + (size_t)l * INNER_, qbuf, nullptr, 256);
    k_gemm<0, 64><<<g4, 128, 0, stream>>>(ybuf, wkT, bk + (size_t)l * INNER_, kbuf, nullptr, 256);
    k_gemm<0, 64><<<g4, 128, 0, stream>>>(ybuf, wvT, bv + (size_t)l * INNER_, vbuf, nullptr, 256);

    k_feat_q<<<BH_ * N_ / 16, 128, 0, stream>>>(qbuf, projS, qpbuf);
    k_feat_kmax<<<BH_ * N_ / 16, 128, 0, stream>>>(kbuf, projS, partial);
    k_rmax<<<1, 256, 0, stream>>>(partial, BH_ * N_ / 16, kmax);
    k_feat_k<<<BH_ * N_ / 16, 128, 0, stream>>>(kbuf, projS, kmax, kpbuf);

    k_kv<<<BH_ * NCH_, 128, 0, stream>>>(kpbuf, vbuf, kvbuf, kscbuf);
    k_prefix_kv<<<(BH_ * DH_ * MP_ + 255) / 256, 256, 0, stream>>>(kvbuf, kvpbuf);
    k_prefix_ks<<<(BH_ * MP_ + 255) / 256, 256, 0, stream>>>(kscbuf, kspbuf);

    k_attn<<<BH_ * NCH_, 256, 0, stream>>>(qpbuf, kpbuf, vbuf, kvpbuf, kspbuf, obuf);
    k_gemm<2, 256><<<g1, 128, 0, stream>>>(obuf, woT, bo + (size_t)l * DIM_, nullptr, hbuf, 64);

    k_ln<<<ROWS_ / 256, 256, 0, stream>>>(hbuf, ln2g + l * DIM_, ln2b + l * DIM_, ybuf);
    k_gemm<1, 64><<<g4, 128, 0, stream>>>(ybuf, w1T, b1 + (size_t)l * INNER_, obuf, nullptr, 256);
    k_gemm<2, 256><<<g1, 128, 0, stream>>>(obuf, w2T, b2 + (size_t)l * DIM_, nullptr, hbuf, 64);
  }

  k_final<<<B_ * NC_, 256, 0, stream>>>(hbuf, Wf, bfv, (float*)d_out);
}
